// EquivariantGNN2D_40407052321388
// MI455X (gfx1250) — matrix operand order repaired and checked by reading, not compiled, not run on hardware
//
#include <hip/hip_runtime.h>
#include <math.h>

typedef _Float16 v16h __attribute__((ext_vector_type(16)));
typedef _Float16 v8h  __attribute__((ext_vector_type(8)));
typedef float    v8f  __attribute__((ext_vector_type(8)));

#define kRS10 0.31622776601683794f
#define kRS3  0.57735026918962584f
#define kRS4  0.5f
#define kRS32 0.17677669529663687f

// path normalizations alpha = sqrt((2*l3+1)/fan_in)
#define kA0 0.24253562503633297f   // sqrt(1/17)
#define kA1 0.36115755925730756f   // sqrt(3/23)
#define kA2 0.48038446141526137f   // sqrt(3/13)
#define kA3 0.45643546458763845f   // sqrt(5/24)

#define kI3 0.57735026918962584f   // 1/sqrt3
#define kI5 0.44721359549995793f   // 1/sqrt5
#define kI6 0.40824829046386302f   // 1/sqrt6

__device__ __forceinline__ float sigmf(float x){ return 1.f/(1.f+__expf(-x)); }

// ---- Wigner-3j contractions in this basis (x,y,z ; xy,yz,q3,xz,x2-y2) ----
__device__ __forceinline__ void cg112(const float* x, const float* y, float* z){
  const float A=0.31622776601683794f, Bc=0.18257418583505536f;
  z[0]=A*(x[0]*y[1]+x[1]*y[0]);
  z[1]=A*(x[1]*y[2]+x[2]*y[1]);
  z[2]=-Bc*(x[0]*y[0]+x[1]*y[1])+2.f*Bc*x[2]*y[2];
  z[3]=A*(x[0]*y[2]+x[2]*y[0]);
  z[4]=A*(x[0]*y[0]-x[1]*y[1]);
}
__device__ __forceinline__ void cg121(const float* x, const float* y, float* z){
  const float A=0.31622776601683794f, Bc=0.18257418583505536f;
  z[0]=A*(x[1]*y[0]+x[2]*y[3]+x[0]*y[4]) - Bc*x[0]*y[2];
  z[1]=A*(x[0]*y[0]+x[2]*y[1]-x[1]*y[4]) - Bc*x[1]*y[2];
  z[2]=A*(x[1]*y[1]+x[0]*y[3]) + 2.f*Bc*x[2]*y[2];
}
__device__ __forceinline__ void cg211(const float* x, const float* y, float* z){
  const float A=0.31622776601683794f, Bc=0.18257418583505536f;
  z[0]=A*(x[0]*y[1]+x[3]*y[2]+x[4]*y[0]) - Bc*x[2]*y[0];
  z[1]=A*(x[0]*y[0]+x[1]*y[2]-x[4]*y[1]) - Bc*x[2]*y[1];
  z[2]=A*(x[1]*y[1]+x[3]*y[0]) + 2.f*Bc*x[2]*y[2];
}
__device__ __forceinline__ void cg122(const float* x, const float* y, float* z){
  const float c=0.18257418583505536f, s3=0.31622776601683794f;
  z[0]= c*x[0]*y[3] - c*x[1]*y[1] - 2.f*c*x[2]*y[4];
  z[1]= s3*x[0]*y[2] + c*x[0]*y[4] + c*x[1]*y[0] - c*x[2]*y[3];
  z[2]= -s3*x[0]*y[1] + s3*x[1]*y[3];
  z[3]= -c*x[0]*y[0] - s3*x[1]*y[2] + c*x[1]*y[4] + c*x[2]*y[1];
  z[4]= -c*x[0]*y[1] - c*x[1]*y[3] + 2.f*c*x[2]*y[0];
}
__device__ __forceinline__ void cg221(const float* x, const float* y, float* z){
  const float c=0.18257418583505536f, s3=0.31622776601683794f;
  z[0]= c*x[3]*y[0] + s3*x[2]*y[1] + c*x[4]*y[1] - s3*x[1]*y[2] - c*x[0]*y[3] - c*x[1]*y[4];
  z[1]= -c*x[1]*y[0] + c*x[0]*y[1] + s3*x[3]*y[2] - s3*x[2]*y[3] + c*x[4]*y[3] - c*x[3]*y[4];
  z[2]= -2.f*c*x[4]*y[0] - c*x[3]*y[1] + c*x[1]*y[3] + 2.f*c*x[0]*y[4];
}
__device__ __forceinline__ void cg222(const float* x, const float* y, float* z){
  const float p=0.2390457218668787f, q=0.11952286093343936f, r=0.20701966780270626f;
  z[0] = p*(x[0]*y[2]+x[2]*y[0]) - r*(x[1]*y[3]+x[3]*y[1]);
  z[1] = -q*(x[1]*y[2]+x[2]*y[1]) + r*(x[1]*y[4]+x[4]*y[1]) - r*(x[0]*y[3]+x[3]*y[0]);
  z[2] = p*x[0]*y[0] - q*x[1]*y[1] - p*x[2]*y[2] - q*x[3]*y[3] + p*x[4]*y[4];
  z[3] = -q*(x[3]*y[2]+x[2]*y[3]) - r*(x[3]*y[4]+x[4]*y[3]) - r*(x[0]*y[1]+x[1]*y[0]);
  z[4] = p*(x[4]*y[2]+x[2]*y[4]) + r*x[1]*y[1] - r*x[3]*y[3];
}

// ---- per-node equivariant linear (10,3x3,3x3,4x5 blocks) ----
__device__ __forceinline__ void lin_hid_dev(const float* h,
    const float* __restrict__ w0, const float* __restrict__ b0,
    const float* __restrict__ w1o, const float* __restrict__ w1e,
    const float* __restrict__ w2e, float* out){
  #pragma unroll
  for(int j=0;j<10;++j){ float s=0.f;
    #pragma unroll
    for(int u=0;u<10;++u) s += h[u]*w0[u*10+j];
    out[j]=s*kRS10+b0[j]; }
  #pragma unroll
  for(int w=0;w<3;++w)
    #pragma unroll
    for(int m=0;m<3;++m){ float s=0.f;
      #pragma unroll
      for(int u=0;u<3;++u) s+=h[10+u*3+m]*w1o[u*3+w];
      out[10+w*3+m]=s*kRS3; }
  #pragma unroll
  for(int w=0;w<3;++w)
    #pragma unroll
    for(int m=0;m<3;++m){ float s=0.f;
      #pragma unroll
      for(int u=0;u<3;++u) s+=h[19+u*3+m]*w1e[u*3+w];
      out[19+w*3+m]=s*kRS3; }
  #pragma unroll
  for(int w=0;w<4;++w)
    #pragma unroll
    for(int m=0;m<5;++m){ float s=0.f;
      #pragma unroll
      for(int u=0;u<4;++u) s+=h[28+u*5+m]*w2e[u*4+w];
      out[28+w*5+m]=s*kRS4; }
}

// ---- edge geometry precompute: rn and SH(9) ----
__global__ void prep_edges(const float* __restrict__ e2, const float* __restrict__ me,
                           float* __restrict__ Yb, float* __restrict__ rnb, int BE){
  int i = blockIdx.x*blockDim.x+threadIdx.x; if(i>=BE) return;
  float m = me[i];
  float ry = e2[i*2+0]*m, rz = e2[i*2+1]*m;
  float rn = sqrtf(ry*ry+rz*rz) + 1e-8f;
  rnb[i]=rn;
  float ux=0.f, uy=ry/rn, uz=rz/rn;
  float* Y = Yb+(size_t)i*9;
  Y[0]=1.f;
  Y[1]=1.7320508075688772f*ux; Y[2]=1.7320508075688772f*uy; Y[3]=1.7320508075688772f*uz;
  Y[4]=3.8729833462074170f*ux*uy; Y[5]=3.8729833462074170f*uy*uz;
  Y[6]=1.1180339887498949f*(2.f*uz*uz-ux*ux-uy*uy);
  Y[7]=3.8729833462074170f*ux*uz;
  Y[8]=1.9364916731037085f*(ux*ux-uy*uy);
}

// ---- pack rad_w2 (L,32,374) into WMMA B-fragments, f16, 24 tiles of 16 cols ----
__global__ void prep_bfrag(const float* __restrict__ w2, _Float16* __restrict__ bf, int Lnum){
  int id = blockIdx.x*blockDim.x+threadIdx.x;
  int tot = Lnum*24*32; if(id>=tot) return;
  int i = id/(24*32); int t = (id/32)%24; int l = id%32;
  int n = t*16 + (l&15);
  int kb = (l<16)?0:16;
  _Float16* dst = bf + (size_t)id*16;
  #pragma unroll
  for(int j=0;j<16;++j){
    int k = kb+j;
    float v = (n<374) ? w2[((size_t)i*32+k)*374 + n] : 0.f;
    dst[j] = (_Float16)v;
  }
}

// ---- encoder: x9 -> typed9 -> h(48) ----
__global__ void encode_kernel(const float* __restrict__ xn, const float* __restrict__ mask,
    const float* __restrict__ w0, const float* __restrict__ b0,
    const float* __restrict__ w1e, const float* __restrict__ w2e,
    float* __restrict__ h, int BN){
  int n = blockIdx.x*blockDim.x+threadIdx.x; if(n>=BN) return;
  const float* x = xn+(size_t)n*9;
  float S00=x[0], S11=x[4], S22=x[8];
  float S01=0.5f*(x[1]+x[3]), S02=0.5f*(x[2]+x[6]), S12=0.5f*(x[5]+x[7]);
  float W12=0.5f*(x[5]-x[7]);   // W[1][2]
  float W20=0.5f*(x[6]-x[2]);   // W[2][0]
  float W01=0.5f*(x[1]-x[3]);   // W[0][1]
  float tr = S00+S11+S22;
  float a[5];
  a[0]=(2.f*S22-S00-S11)*0.40824829046386302f;
  a[1]=(S00-S11)*0.70710678118654752f;
  a[2]=1.4142135623730951f*S01;
  a[3]=1.4142135623730951f*S02;
  a[4]=1.4142135623730951f*S12;
  float om[3]={W12,W20,W01};
  float mk = mask[n];
  float* hp=h+(size_t)n*48;
  #pragma unroll
  for(int j=0;j<10;++j) hp[j]=(tr*w0[j]+b0[j])*mk;
  #pragma unroll
  for(int i=0;i<9;++i) hp[10+i]=0.f;
  #pragma unroll
  for(int w=0;w<3;++w)
    #pragma unroll
    for(int m=0;m<3;++m) hp[19+w*3+m]=om[m]*w1e[w]*mk;
  #pragma unroll
  for(int w=0;w<4;++w)
    #pragma unroll
    for(int m=0;m<5;++m) hp[28+w*5+m]=a[m]*w2e[w]*mk;
}

__global__ void linhid_kernel(const float* __restrict__ h,
    const float* __restrict__ w0,const float* __restrict__ b0,
    const float* __restrict__ w1o,const float* __restrict__ w1e,
    const float* __restrict__ w2e, float* __restrict__ hout,int BN){
  int n = blockIdx.x*blockDim.x+threadIdx.x; if(n>=BN) return;
  float hv[48];
  #pragma unroll
  for(int i=0;i<48;++i) hv[i]=h[(size_t)n*48+i];
  float ov[48];
  lin_hid_dev(hv,w0,b0,w1o,w1e,w2e,ov);
  #pragma unroll
  for(int i=0;i<48;++i) hout[(size_t)n*48+i]=ov[i];
}

__global__ void zero_kernel(float* p, int n){
  int i = blockIdx.x*blockDim.x+threadIdx.x; if(i<n) p[i]=0.f;
}

// ---- fused edge kernel: radial MLP via WMMA + CG tensor product + scatter ----
__global__ __launch_bounds__(32) void edge_kernel(
    const float* __restrict__ hin, const float* __restrict__ Yb,
    const float* __restrict__ rnb, const float* __restrict__ w1,
    const _Float16* __restrict__ bfrag,
    const int* __restrict__ e_src, const int* __restrict__ e_dst,
    float* __restrict__ agg, int BE, int E, int N)
{
  __shared__ __attribute__((aligned(32))) _Float16 hid16[16*32];   // 16 edges x 32 hidden
  __shared__ __attribute__((aligned(32))) _Float16 wlds[32*384];   // 32 edges x 384 weights
  __shared__ float rns[32];
  const int lane = threadIdx.x;
  const int eg = blockIdx.x*32 + lane;
  rns[lane] = (eg<BE)? rnb[eg] : 0.f;
  const float w1l = w1[lane];
  __syncthreads();

  const int Mrow = lane & 15;
  const int koff = (lane>>4)*8;

  // -------- phase 1: W[32 edges][374] = silu(rn*w1) @ rad_w2 * RS32 via WMMA --------
  for(int g=0; g<2; ++g){
    #pragma unroll 4
    for(int e2=0;e2<16;++e2){
      float rv = rns[g*16+e2];
      float t  = rv*w1l;
      hid16[e2*32+lane] = (_Float16)(t*sigmf(t));   // silu
    }
    __syncthreads();
    v16h a;
    { v8h lo = *(const v8h*)&hid16[Mrow*32+koff*2];
      v8h hi = *(const v8h*)&hid16[Mrow*32+koff*2+8];
      #pragma unroll
      for(int j=0;j<8;++j){ a[j]=lo[j]; a[8+j]=hi[j]; } }
    for(int t=0;t<24;++t){
      v16h bvec = *(const v16h*)&bfrag[(size_t)(t*32+lane)*16];
      v8f c = {};
      c = __builtin_amdgcn_wmma_f32_16x16x32_f16(false, a, false, bvec,
                                                 (short)0, c, false, false);
      int col = t*16 + Mrow;
      #pragma unroll
      for(int r2=0;r2<8;++r2)
        wlds[(size_t)(g*16+koff+r2)*384 + col] = (_Float16)(c[r2]*kRS32);
    }
    __syncthreads();
  }

  // -------- phase 2: per-lane tensor product + atomic scatter --------
  if(eg >= BE) return;
  int b   = eg / E;
  int src = e_src[eg] + b*N;
  int dst = e_dst[eg] + b*N;
  float hs[48];
  { const float* hp = hin + (size_t)src*48;
    #pragma unroll
    for(int i=0;i<48;++i) hs[i]=hp[i]; }
  const float* Yp = Yb + (size_t)eg*9;
  float y0=Yp[0];
  float y1[3]={Yp[1],Yp[2],Yp[3]};
  float y2[5]={Yp[4],Yp[5],Yp[6],Yp[7],Yp[8]};
  const _Float16* wl = &wlds[(size_t)lane*384];
  auto W=[&](int j)->float{ return (float)wl[j]; };

  float o0[10]={}, o1[9]={}, o2[9]={}, o3[20]={};

  // P0 (0,0,0)->0  off 0, 10x10
  { float f=kA0*y0;
    #pragma unroll
    for(int u=0;u<10;++u){ float xv=f*hs[u];
      #pragma unroll
      for(int w=0;w<10;++w) o0[w]+=W(u*10+w)*xv; } }
  // P1 (0,1,1)->1  off 100, 10x3
  { float f=kA1*kI3;
    #pragma unroll
    for(int u=0;u<10;++u){ float xv=f*hs[u];
      #pragma unroll
      for(int w=0;w<3;++w){ float t=W(100+u*3+w)*xv;
        #pragma unroll
        for(int k=0;k<3;++k) o1[w*3+k]+=t*y1[k]; } } }
  // P2 (0,2,2)->3  off 130, 10x4
  { float f=kA3*kI5;
    #pragma unroll
    for(int u=0;u<10;++u){ float xv=f*hs[u];
      #pragma unroll
      for(int w=0;w<4;++w){ float t=W(130+u*4+w)*xv;
        #pragma unroll
        for(int k=0;k<5;++k) o3[w*5+k]+=t*y2[k]; } } }
  // P3 (1,0,1)->1  off 170, 3x3
  { float f=kA1*kI3*y0;
    #pragma unroll
    for(int u=0;u<3;++u)
      #pragma unroll
      for(int w=0;w<3;++w){ float t=W(170+u*3+w)*f;
        #pragma unroll
        for(int k=0;k<3;++k) o1[w*3+k]+=t*hs[10+u*3+k]; } }
  // P4 (1,1,0)->0  off 179, 3x10
  { float f=kA0*kI3;
    #pragma unroll
    for(int u=0;u<3;++u){ const float* x=&hs[10+u*3];
      float zu=f*(x[0]*y1[0]+x[1]*y1[1]+x[2]*y1[2]);
      #pragma unroll
      for(int w=0;w<10;++w) o0[w]+=W(179+u*10+w)*zu; } }
  // P5 (1,1,1)->2  off 209, 3x3  (cross / eps)
  { float f=kA2*kI6;
    #pragma unroll
    for(int u=0;u<3;++u){ const float* x=&hs[10+u*3];
      float z0=f*(x[1]*y1[2]-x[2]*y1[1]);
      float z1=f*(x[2]*y1[0]-x[0]*y1[2]);
      float z2=f*(x[0]*y1[1]-x[1]*y1[0]);
      #pragma unroll
      for(int w=0;w<3;++w){ float wp=W(209+u*3+w);
        o2[w*3+0]+=wp*z0; o2[w*3+1]+=wp*z1; o2[w*3+2]+=wp*z2; } } }
  // P6 (1,1,2)->3  off 218, 3x4
  {
    #pragma unroll
    for(int u=0;u<3;++u){ float z[5]; cg112(&hs[10+u*3], y1, z);
      #pragma unroll
      for(int w=0;w<4;++w){ float wp=kA3*W(218+u*4+w);
        #pragma unroll
        for(int k=0;k<5;++k) o3[w*5+k]+=wp*z[k]; } }
  }
  // P7 (1,2,1)->1  off 230, 3x3
  {
    #pragma unroll
    for(int u=0;u<3;++u){ float z[3]; cg121(&hs[10+u*3], y2, z);
      #pragma unroll
      for(int w=0;w<3;++w){ float wp=kA1*W(230+u*3+w);
        #pragma unroll
        for(int k=0;k<3;++k) o1[w*3+k]+=wp*z[k]; } }
  }
  // P8 (1,0,1)->2  off 239, 3x3
  { float f=kA2*kI3*y0;
    #pragma unroll
    for(int u=0;u<3;++u)
      #pragma unroll
      for(int w=0;w<3;++w){ float t=W(239+u*3+w)*f;
        #pragma unroll
        for(int k=0;k<3;++k) o2[w*3+k]+=t*hs[19+u*3+k]; } }
  // P9 (1,1,1)->1  off 248, 3x3
  { float f=kA1*kI6;
    #pragma unroll
    for(int u=0;u<3;++u){ const float* x=&hs[19+u*3];
      float z0=f*(x[1]*y1[2]-x[2]*y1[1]);
      float z1=f*(x[2]*y1[0]-x[0]*y1[2]);
      float z2=f*(x[0]*y1[1]-x[1]*y1[0]);
      #pragma unroll
      for(int w=0;w<3;++w){ float wp=W(248+u*3+w);
        o1[w*3+0]+=wp*z0; o1[w*3+1]+=wp*z1; o1[w*3+2]+=wp*z2; } } }
  // P10 (1,2,1)->2 off 257, 3x3
  {
    #pragma unroll
    for(int u=0;u<3;++u){ float z[3]; cg121(&hs[19+u*3], y2, z);
      #pragma unroll
      for(int w=0;w<3;++w){ float wp=kA2*W(257+u*3+w);
        #pragma unroll
        for(int k=0;k<3;++k) o2[w*3+k]+=wp*z[k]; } }
  }
  // P11 (1,2,2)->3 off 266, 3x4
  {
    #pragma unroll
    for(int u=0;u<3;++u){ float z[5]; cg122(&hs[19+u*3], y2, z);
      #pragma unroll
      for(int w=0;w<4;++w){ float wp=kA3*W(266+u*4+w);
        #pragma unroll
        for(int k=0;k<5;++k) o3[w*5+k]+=wp*z[k]; } }
  }
  // P12 (2,0,2)->3 off 278, 4x4
  { float f=kA3*kI5*y0;
    #pragma unroll
    for(int u=0;u<4;++u)
      #pragma unroll
      for(int w=0;w<4;++w){ float t=W(278+u*4+w)*f;
        #pragma unroll
        for(int k=0;k<5;++k) o3[w*5+k]+=t*hs[28+u*5+k]; } }
  // P13 (2,1,1)->1 off 294, 4x3
  {
    #pragma unroll
    for(int u=0;u<4;++u){ float z[3]; cg211(&hs[28+u*5], y1, z);
      #pragma unroll
      for(int w=0;w<3;++w){ float wp=kA1*W(294+u*3+w);
        #pragma unroll
        for(int k=0;k<3;++k) o1[w*3+k]+=wp*z[k]; } }
  }
  // P14 (2,2,0)->0 off 306, 4x10
  { float f=kA0*kI5;
    #pragma unroll
    for(int u=0;u<4;++u){ const float* x=&hs[28+u*5];
      float zu=f*(x[0]*y2[0]+x[1]*y2[1]+x[2]*y2[2]+x[3]*y2[3]+x[4]*y2[4]);
      #pragma unroll
      for(int w=0;w<10;++w) o0[w]+=W(306+u*10+w)*zu; } }
  // P15 (2,2,1)->2 off 346, 4x3
  {
    #pragma unroll
    for(int u=0;u<4;++u){ float z[3]; cg221(&hs[28+u*5], y2, z);
      #pragma unroll
      for(int w=0;w<3;++w){ float wp=kA2*W(346+u*3+w);
        #pragma unroll
        for(int k=0;k<3;++k) o2[w*3+k]+=wp*z[k]; } }
  }
  // P16 (2,2,2)->3 off 358, 4x4
  {
    #pragma unroll
    for(int u=0;u<4;++u){ float z[5]; cg222(&hs[28+u*5], y2, z);
      #pragma unroll
      for(int w=0;w<4;++w){ float wp=kA3*W(358+u*4+w);
        #pragma unroll
        for(int k=0;k<5;++k) o3[w*5+k]+=wp*z[k]; } }
  }

  float* ap = agg + (size_t)dst*48;
  #pragma unroll
  for(int i=0;i<10;++i) atomicAdd(ap+i, o0[i]);
  #pragma unroll
  for(int i=0;i<9;++i)  atomicAdd(ap+10+i, o1[i]);
  #pragma unroll
  for(int i=0;i<9;++i)  atomicAdd(ap+19+i, o2[i]);
  #pragma unroll
  for(int i=0;i<20;++i) atomicAdd(ap+28+i, o3[i]);
}

// ---- combine: h' = norm_act(agg + lin_hid_lr(h)) * mask ----
template<int OFF,int M,int D>
__device__ __forceinline__ void norm_act_block(const float* v, float* out){
  #pragma unroll
  for(int r=0;r<M;++r){
    float ss=1e-12f;
    #pragma unroll
    for(int k=0;k<D;++k){ float x=v[OFF+r*D+k]; ss+=x*x; }
    float sg = sigmf(sqrtf(ss));
    #pragma unroll
    for(int k=0;k<D;++k) out[OFF+r*D+k]=v[OFF+r*D+k]*sg;
  }
}

__global__ void combine_kernel(const float* __restrict__ h, const float* __restrict__ agg,
    const float* __restrict__ mask,
    const float* __restrict__ w0,const float* __restrict__ b0,
    const float* __restrict__ w1o,const float* __restrict__ w1e,
    const float* __restrict__ w2e, float* __restrict__ hout,int BN){
  int n = blockIdx.x*blockDim.x+threadIdx.x; if(n>=BN) return;
  float hv[48];
  #pragma unroll
  for(int i=0;i<48;++i) hv[i]=h[(size_t)n*48+i];
  float v[48];
  lin_hid_dev(hv,w0,b0,w1o,w1e,w2e,v);
  #pragma unroll
  for(int i=0;i<48;++i) v[i]+=agg[(size_t)n*48+i];
  float out[48];
  norm_act_block<0,10,1>(v,out);
  norm_act_block<10,3,3>(v,out);
  norm_act_block<19,3,3>(v,out);
  norm_act_block<28,4,5>(v,out);
  float mk = mask[n];
  #pragma unroll
  for(int i=0;i<48;++i) hout[(size_t)n*48+i]=out[i]*mk;
}

// ---- readout at center nodes ----
__global__ void output_kernel(const float* __restrict__ h, const int* __restrict__ centers,
    const float* __restrict__ sw0, const float* __restrict__ sb0,
    const float* __restrict__ sw2, float* __restrict__ out, int Bn, int N){
  int b = threadIdx.x; if(b>=Bn) return;
  int c = centers[b];
  const float* hc = h + ((size_t)b*N + (size_t)c)*48;
  float acc=0.f;
  #pragma unroll
  for(int u=0;u<10;++u) acc += hc[u]*sw0[u];
  float tr = acc*kRS10 + sb0[0];
  float s2[5];
  #pragma unroll
  for(int m=0;m<5;++m){ float s=0.f;
    #pragma unroll
    for(int u=0;u<4;++u) s+=hc[28+u*5+m]*sw2[u];
    s2[m]=s*kRS4; }
  float a0=s2[0], a2c=s2[1], a2s=s2[2], a1c=s2[3], a1s=s2[4];
  const float is6=0.40824829046386302f, is2=0.70710678118654752f;
  out[b*6+0]=-a0*is6 + a2c*is2 + tr/3.f;
  out[b*6+1]=a2s;
  out[b*6+2]=a1c;
  out[b*6+3]=-a0*is6 - a2c*is2 + tr/3.f;
  out[b*6+4]=a1s;
  out[b*6+5]=2.f*a0*is6 + tr/3.f;
}

extern "C" void kernel_launch(void* const* d_in, const int* in_sizes, int n_in,
                              void* d_out, int out_size, void* d_ws, size_t ws_size,
                              hipStream_t stream){
  (void)n_in; (void)out_size; (void)ws_size;
  const float* x_nodes   =(const float*)d_in[0];
  const float* e_attr2   =(const float*)d_in[1];
  const float* mask_nodes=(const float*)d_in[2];
  const float* mask_edges=(const float*)d_in[3];
  const float* enc_w0 =(const float*)d_in[4];
  const float* enc_b0 =(const float*)d_in[5];
  const float* enc_w1e=(const float*)d_in[6];
  const float* enc_w2e=(const float*)d_in[7];
  const float* li_w0 =(const float*)d_in[8];
  const float* li_b0 =(const float*)d_in[9];
  const float* li_w1o=(const float*)d_in[10];
  const float* li_w1e=(const float*)d_in[11];
  const float* li_w2e=(const float*)d_in[12];
  const float* rad_w1=(const float*)d_in[13];
  const float* rad_w2=(const float*)d_in[14];
  const float* lr_w0 =(const float*)d_in[15];
  const float* lr_b0 =(const float*)d_in[16];
  const float* lr_w1o=(const float*)d_in[17];
  const float* lr_w1e=(const float*)d_in[18];
  const float* lr_w2e=(const float*)d_in[19];
  const float* sym_w0 =(const float*)d_in[20];
  const float* sym_b0 =(const float*)d_in[21];
  const float* sym_w2e=(const float*)d_in[22];
  const int* e_src  =(const int*)d_in[23];
  const int* e_dst  =(const int*)d_in[24];
  const int* centers=(const int*)d_in[25];
  float* out=(float*)d_out;

  const int Bn = in_sizes[25];
  const int BE = in_sizes[23];
  const int E  = BE / Bn;
  const int BN = in_sizes[0]/9;
  const int N  = BN / Bn;
  const int L  = in_sizes[8]/100;

  float* ws = (float*)d_ws;
  size_t o=0;
  float* hA  = ws+o; o += (size_t)BN*48;
  float* hB  = ws+o; o += (size_t)BN*48;
  float* hin = ws+o; o += (size_t)BN*48;
  float* agg = ws+o; o += (size_t)BN*48;
  float* Yb  = ws+o; o += (size_t)BE*9;
  float* rnb = ws+o; o += (size_t)BE;
  o = (o+15)&~(size_t)15;
  _Float16* bfrag = (_Float16*)(ws+o);

  prep_edges<<<(BE+255)/256,256,0,stream>>>(e_attr2, mask_edges, Yb, rnb, BE);
  int nbf = L*24*32;
  prep_bfrag<<<(nbf+255)/256,256,0,stream>>>(rad_w2, bfrag, L);
  encode_kernel<<<(BN+255)/256,256,0,stream>>>(x_nodes, mask_nodes, enc_w0, enc_b0,
                                               enc_w1e, enc_w2e, hA, BN);

  float* cur=hA; float* nxt=hB;
  for(int l=0;l<L;++l){
    linhid_kernel<<<(BN+255)/256,256,0,stream>>>(cur, li_w0+(size_t)l*100, li_b0+(size_t)l*10,
        li_w1o+(size_t)l*9, li_w1e+(size_t)l*9, li_w2e+(size_t)l*16, hin, BN);
    int ntot = BN*48;
    zero_kernel<<<(ntot+255)/256,256,0,stream>>>(agg, ntot);
    edge_kernel<<<(BE+31)/32,32,0,stream>>>(hin, Yb, rnb, rad_w1+(size_t)l*32,
        bfrag+(size_t)l*24*32*16, e_src, e_dst, agg, BE, E, N);
    combine_kernel<<<(BN+255)/256,256,0,stream>>>(cur, agg, mask_nodes,
        lr_w0+(size_t)l*100, lr_b0+(size_t)l*10, lr_w1o+(size_t)l*9,
        lr_w1e+(size_t)l*9, lr_w2e+(size_t)l*16, nxt, BN);
    float* tswap=cur; cur=nxt; nxt=tswap;
  }
  output_kernel<<<1,Bn,0,stream>>>(cur, centers, sym_w0, sym_b0, sym_w2e, out, Bn, N);
}
